// MultiScaleEdgeConv_36567351558298
// MI455X (gfx1250) — matrix operand order repaired and checked by reading, not compiled, not run on hardware
//
#include <hip/hip_runtime.h>
#include <hip/hip_bf16.h>

typedef __attribute__((ext_vector_type(16))) __bf16 v16bf;
typedef __attribute__((ext_vector_type(8)))  float  v8f;
typedef unsigned short u16t;
typedef unsigned int   u32t;

#define Bsz 128
#define Pn  128
#define Kn  16
#define NODE_DIM 64
#define MSG_DIM 128
#define OUT_DIM 128
#define EDGE_IN 132
#define EDGE_CPAD 160      // padded channels (5 chunks of 32)

#define ACT1_STRIDE 168    // ushorts per edge row (pad: 84 dwords -> bank spread)
#define ACT2_STRIDE 136    // ushorts per row for 128-channel activations

static constexpr float PI_F    = 3.14159265358979323846f;
static constexpr float TWO_PI  = 6.283185307179586f;
static constexpr float INV2PI  = 0.15915494309189535f;

// ---------- workspace layout (bytes) ----------
static constexpr size_t OFF_IDX = 0;                                   // B*P*K int
static constexpr size_t SZ_IDX  = (size_t)Bsz*Pn*Kn*4;
static constexpr size_t OFF_MSG = OFF_IDX + SZ_IDX;                    // B*128*P f32
static constexpr size_t SZ_MSG  = (size_t)Bsz*MSG_DIM*Pn*4;
static constexpr size_t OFF_W1  = OFF_MSG + SZ_MSG;                    // 128x160 bf16
static constexpr size_t SZ_W1   = (size_t)MSG_DIM*EDGE_CPAD*2;
static constexpr size_t OFF_W2  = OFF_W1 + SZ_W1;                      // 128x128 bf16
static constexpr size_t SZ_W2   = (size_t)MSG_DIM*MSG_DIM*2;
static constexpr size_t OFF_NW1 = OFF_W2 + SZ_W2;
static constexpr size_t OFF_NW2 = OFF_NW1 + SZ_W2;
static constexpr size_t OFF_SCB = OFF_NW2 + SZ_W2;                     // 8 * 160 floats

__device__ __forceinline__ u16t f2bf(float f) {
  u32t u = __float_as_uint(f);
  u += 0x7FFFu + ((u >> 16) & 1u);           // round-to-nearest-even
  return (u16t)(u >> 16);
}

// ---- WMMA fragment loads per CDNA5 ISA 7.12.2 ----
// A matrix 16x32 bf16: lane m (0..15)/m+16; K = half*8 + {0..7} and {16..23}
__device__ __forceinline__ v16bf load_a_frag(const u16t* __restrict__ W, int stride,
                                             int row0, int kc, int lane) {
  int m = lane & 15, half = lane >> 4;
  const u16t* p = W + (size_t)(row0 + m) * stride + kc * 32 + half * 8;
  union { v16bf v; uint4 q[2]; } u;
  u.q[0] = *(const uint4*)p;         // K = half*8 + 0..7
  u.q[1] = *(const uint4*)(p + 16);  // K = half*8 + 16..23
  return u.v;
}
// B matrix 32x16 bf16: lane n holds column n; K = half*8 + {0..7} and {16..23}
__device__ __forceinline__ v16bf load_b_frag(const u16t* act, int stride,
                                             int etile, int kc, int lane) {
  int n = lane & 15, half = lane >> 4;
  const u16t* p = act + (size_t)(etile * 16 + n) * stride + kc * 32 + half * 8;
  union { v16bf v; uint4 q[2]; } u;
  u.q[0] = *(const uint4*)p;
  u.q[1] = *(const uint4*)(p + 16);
  return u.v;
}
// BN+ReLU an accumulator half-column (8 rows) and pack to 8 bf16 (one b128 store)
__device__ __forceinline__ uint4 pack_bn_relu(v8f a, const float* s, const float* t, int cbase) {
  u32t r[4];
#pragma unroll
  for (int i = 0; i < 4; ++i) {
    float y0 = fmaxf(a[2*i+0] * s[cbase + 2*i+0] + t[cbase + 2*i+0], 0.f);
    float y1 = fmaxf(a[2*i+1] * s[cbase + 2*i+1] + t[cbase + 2*i+1], 0.f);
    r[i] = (u32t)f2bf(y0) | ((u32t)f2bf(y1) << 16);
  }
  return make_uint4(r[0], r[1], r[2], r[3]);
}

__device__ __forceinline__ float wrap_dphi(float x) {
  float a = x + PI_F;
  a -= floorf(a * INV2PI) * TWO_PI;
  return a - PI_F;
}

// ============================ kernel 0: prep ============================
__global__ void prep_kernel(const float* eg1, const float* eb1, const float* em1, const float* ev1,
                            const float* ew1,
                            const float* eg2, const float* eb2, const float* em2, const float* ev2,
                            const float* ew2,
                            const float* ng1, const float* nb1, const float* nm1, const float* nv1,
                            const float* nw1,
                            const float* ng2, const float* nb2, const float* nm2, const float* nv2,
                            const float* nw2,
                            u16t* w1bf, u16t* w2bf, u16t* nw1bf, u16t* nw2bf, float* scb) {
  int t = threadIdx.x;
  if (t < EDGE_IN) { float s = eg1[t] * rsqrtf(ev1[t] + 1e-5f); scb[t] = s; scb[160 + t] = eb1[t] - em1[t] * s; }
  if (t < MSG_DIM) {
    float s2 = eg2[t] * rsqrtf(ev2[t] + 1e-5f); scb[320 + t] = s2; scb[480 + t] = eb2[t] - em2[t] * s2;
    float s3 = ng1[t] * rsqrtf(nv1[t] + 1e-5f); scb[640 + t] = s3; scb[800 + t] = nb1[t] - nm1[t] * s3;
    float s4 = ng2[t] * rsqrtf(nv2[t] + 1e-5f); scb[960 + t] = s4; scb[1120 + t] = nb2[t] - nm2[t] * s4;
  }
  for (int i = t; i < MSG_DIM * EDGE_CPAD; i += blockDim.x) {
    int o = i / EDGE_CPAD, c = i % EDGE_CPAD;
    w1bf[i] = (c < EDGE_IN) ? f2bf(ew1[o * EDGE_IN + c]) : (u16t)0;
  }
  for (int i = t; i < MSG_DIM * MSG_DIM; i += blockDim.x) {
    w2bf[i]  = f2bf(ew2[i]);
    nw1bf[i] = f2bf(nw1[i]);
    nw2bf[i] = f2bf(nw2[i]);
  }
}

// ============================ kernel 1: KNN ============================
// one wave per (b,p): lane L owns candidates j = 4L..4L+3
__global__ void knn_kernel(const float* __restrict__ pts, const unsigned char* __restrict__ mask,
                           int* __restrict__ idx) {
  int gw   = (blockIdx.x * blockDim.x + threadIdx.x) >> 5;
  int lane = threadIdx.x & 31;
  int b = gw >> 7, p = gw & 127;
  const float* eta = pts + (size_t)b * 2 * Pn;
  const float* phi = eta + Pn;
  const unsigned char* mk = mask + (size_t)b * Pn;
  float ei = eta[p] + (mk[p] ? 0.f : 1e6f);
  float pf = phi[p];
  int jb = lane * 4;
  float d[4];
#pragma unroll
  for (int q = 0; q < 4; ++q) {
    int j = jb + q;
    float ej = eta[j] + (mk[j] ? 0.f : 1e6f);
    float de = ei - ej;
    float dp = wrap_dphi(pf - phi[j]);
    d[q] = de * de + dp * dp;
  }
  int mine = 0;
#pragma unroll
  for (int k = 0; k < Kn; ++k) {
    float bv = d[0]; int bi = jb;
#pragma unroll
    for (int q = 1; q < 4; ++q) { if (d[q] < bv) { bv = d[q]; bi = jb + q; } }
#pragma unroll
    for (int off = 16; off >= 1; off >>= 1) {
      float ov = __shfl_xor(bv, off, 32);
      int   oi = __shfl_xor(bi, off, 32);
      if (ov < bv || (ov == bv && oi < bi)) { bv = ov; bi = oi; }
    }
    if (lane == k) mine = bi;
    if ((bi >> 2) == lane) {                 // retire the winner
      int q = bi & 3;
      d[0] = (q == 0) ? 3e38f : d[0];
      d[1] = (q == 1) ? 3e38f : d[1];
      d[2] = (q == 2) ? 3e38f : d[2];
      d[3] = (q == 3) ? 3e38f : d[3];
    }
  }
  if (lane < Kn) idx[((size_t)b * Pn + p) * Kn + lane] = mine;
}

// ===================== kernel 2: edge MLP + masked max =====================
// grid: B * (P/8); workgroup = 256 threads (8 waves); wave w owns point pl=w
__global__ void __launch_bounds__(256)
edge_mlp_kernel(const float* __restrict__ lvs, const float* __restrict__ fts,
                const unsigned char* __restrict__ mask, const int* __restrict__ idx,
                const u16t* __restrict__ w1bf, const u16t* __restrict__ w2bf,
                const float* __restrict__ scb, float* __restrict__ msg) {
  __shared__ u16t act[128 * ACT1_STRIDE];   // reused: stage1 [e][160], stage2 [e][128]
  __shared__ int  idxLds[128];
  __shared__ unsigned char nullLds[128];

  const float* s1 = scb;        const float* t1 = scb + 160;
  const float* s2 = scb + 320;  const float* t2 = scb + 480;

  int b  = blockIdx.x >> 4;
  int p0 = (blockIdx.x & 15) * 8;
  int tid = threadIdx.x;

  // ---- phase A: per-edge pairwise LV features + null flags ----
  if (tid < 128) {
    int e = tid, pl = e >> 4, k = e & 15;
    int p = p0 + pl;
    int j = idx[((size_t)b * Pn + p) * Kn + k];
    idxLds[e] = j;
    nullLds[e] = (mask[(size_t)b * Pn + p] && mask[(size_t)b * Pn + j]) ? 0 : 1;
    const float* lv = lvs + (size_t)b * 4 * Pn;
    float xi0 = lv[0*Pn + p], xi1 = lv[1*Pn + p], xi2 = lv[2*Pn + p], xi3 = lv[3*Pn + p];
    float xj0 = lv[0*Pn + j], xj1 = lv[1*Pn + j], xj2 = lv[2*Pn + j], xj3 = lv[3*Pn + j];
    float pti = sqrtf(fmaxf(xi0*xi0 + xi1*xi1, 1e-16f));
    float ptj = sqrtf(fmaxf(xj0*xj0 + xj1*xj1, 1e-16f));
    float rapi = 0.5f * logf(1.f + 2.f * xi2 / fmaxf(xi3 - xi2, 1e-20f));
    float rapj = 0.5f * logf(1.f + 2.f * xj2 / fmaxf(xj3 - xj2, 1e-20f));
    float dphi = wrap_dphi(atan2f(xi1, xi0) - atan2f(xj1, xj0));
    float drap = rapi - rapj;
    float delta = sqrtf(fmaxf(drap * drap + dphi * dphi, 1e-16f));
    float ptmin = fminf(pti, ptj);
    float lnkt = logf(fmaxf(ptmin * delta, 1e-8f));
    float lnz  = logf(fmaxf(ptmin / fmaxf(pti + ptj, 1e-8f), 1e-8f));
    float lnd  = logf(fmaxf(delta, 1e-8f));
    float sx = xi0 + xj0, sy = xi1 + xj1, sz = xi2 + xj2, se = xi3 + xj3;
    float lnm2 = logf(fmaxf(se * se - (sx * sx + sy * sy + sz * sz), 1e-8f));
    float f[4] = { lnkt, lnz, lnd, lnm2 };
#pragma unroll
    for (int c = 0; c < 4; ++c) {
      int ch = 128 + c;
      act[e * ACT1_STRIDE + ch] = f2bf(fmaxf(f[c] * s1[ch] + t1[ch], 0.f));
    }
  }
  __syncthreads();

  // ---- phase B: feature channels (center 0..63, neighbor 64..127) + zero pad ----
  const float* fb = fts + (size_t)b * NODE_DIM * Pn;
  for (int i = tid; i < 128 * 128; i += 256) {
    int c = i >> 7, e = i & 127;
    int p = p0 + (e >> 4);
    float x = (c < NODE_DIM) ? fb[c * Pn + p] : fb[(c - NODE_DIM) * Pn + idxLds[e]];
    act[e * ACT1_STRIDE + c] = f2bf(fmaxf(x * s1[c] + t1[c], 0.f));
  }
  for (int i = tid; i < 128 * (EDGE_CPAD - EDGE_IN); i += 256) {
    int c = EDGE_IN + (i >> 7), e = i & 127;
    act[e * ACT1_STRIDE + c] = 0;
  }
  __syncthreads();

  int w = tid >> 5, lane = tid & 31, n = lane & 15, half = lane >> 4;
  v8f zero = { 0.f, 0.f, 0.f, 0.f, 0.f, 0.f, 0.f, 0.f };

  // ---- GEMM1: [128 out] x [160 ch] x [16 edges of point w] ----
  v8f acc[8];
#pragma unroll
  for (int ot = 0; ot < 8; ++ot) acc[ot] = zero;
#pragma unroll
  for (int kc = 0; kc < 5; ++kc) {
    v16bf bf = load_b_frag(act, ACT1_STRIDE, w, kc, lane);
#pragma unroll
    for (int ot = 0; ot < 8; ++ot) {
      v16bf af = load_a_frag(w1bf, EDGE_CPAD, ot * 16, kc, lane);
      acc[ot] = __builtin_amdgcn_wmma_f32_16x16x32_bf16(false, af, false, bf,
                                                        (short)0, acc[ot], false, false);
    }
  }
  __syncthreads();                       // all reads of act done -> safe to alias

  // ---- BN2 + ReLU -> bf16 repack into same LDS (stride 136) ----
#pragma unroll
  for (int ot = 0; ot < 8; ++ot) {
    int cb = ot * 16 + half * 8;
    uint4 q = pack_bn_relu(acc[ot], s2, t2, cb);
    *(uint4*)&act[(size_t)(w * 16 + n) * ACT2_STRIDE + cb] = q;
  }
  __syncthreads();

  // ---- GEMM2: [128 out] x [128 ch] ----
  v8f acc2[8];
#pragma unroll
  for (int ot = 0; ot < 8; ++ot) acc2[ot] = zero;
#pragma unroll
  for (int kc = 0; kc < 4; ++kc) {
    v16bf bf = load_b_frag(act, ACT2_STRIDE, w, kc, lane);
#pragma unroll
    for (int ot = 0; ot < 8; ++ot) {
      v16bf af = load_a_frag(w2bf, MSG_DIM, ot * 16, kc, lane);
      acc2[ot] = __builtin_amdgcn_wmma_f32_16x16x32_bf16(false, af, false, bf,
                                                         (short)0, acc2[ot], false, false);
    }
  }

  // ---- null-edge mask + max over K (columns = neighbor idx) ----
  bool nul = nullLds[w * 16 + n] != 0;
  int p = p0 + w;
#pragma unroll
  for (int ot = 0; ot < 8; ++ot) {
    v8f val = acc2[ot];
#pragma unroll
    for (int v = 0; v < 8; ++v) if (nul) val[v] = -1e9f;
#pragma unroll
    for (int off = 1; off <= 8; off <<= 1) {
#pragma unroll
      for (int v = 0; v < 8; ++v) {
        float o = __shfl_xor(val[v], off, 32);
        val[v] = fmaxf(val[v], o);
      }
    }
    if (n == 0) {
#pragma unroll
      for (int v = 0; v < 8; ++v) {
        int o = ot * 16 + half * 8 + v;
        msg[(size_t)b * MSG_DIM * Pn + (size_t)o * Pn + p] = val[v];
      }
    }
  }
}

// ============================ kernel 3: node MLP ============================
// one workgroup per batch; wave w owns point tile p = 16w..16w+15
__global__ void __launch_bounds__(256)
node_mlp_kernel(const float* __restrict__ msg, const unsigned char* __restrict__ mask,
                const u16t* __restrict__ nw1bf, const u16t* __restrict__ nw2bf,
                const float* __restrict__ scb, float* __restrict__ out) {
  __shared__ u16t act[128 * ACT2_STRIDE];
  const float* sn1 = scb + 640;  const float* tn1 = scb + 800;
  const float* sn2 = scb + 960;  const float* tn2 = scb + 1120;
  int b = blockIdx.x, tid = threadIdx.x;

  for (int i = tid; i < 128 * 128; i += 256) {
    int c = i >> 7, p = i & 127;
    float x = msg[(size_t)b * MSG_DIM * Pn + i];
    act[p * ACT2_STRIDE + c] = f2bf(fmaxf(x * sn1[c] + tn1[c], 0.f));
  }
  __syncthreads();

  int w = tid >> 5, lane = tid & 31, n = lane & 15, half = lane >> 4;
  v8f zero = { 0.f, 0.f, 0.f, 0.f, 0.f, 0.f, 0.f, 0.f };

  v8f acc[8];
#pragma unroll
  for (int ot = 0; ot < 8; ++ot) acc[ot] = zero;
#pragma unroll
  for (int kc = 0; kc < 4; ++kc) {
    v16bf bf = load_b_frag(act, ACT2_STRIDE, w, kc, lane);
#pragma unroll
    for (int ot = 0; ot < 8; ++ot) {
      v16bf af = load_a_frag(nw1bf, MSG_DIM, ot * 16, kc, lane);
      acc[ot] = __builtin_amdgcn_wmma_f32_16x16x32_bf16(false, af, false, bf,
                                                        (short)0, acc[ot], false, false);
    }
  }
  __syncthreads();
#pragma unroll
  for (int ot = 0; ot < 8; ++ot) {
    int cb = ot * 16 + half * 8;
    uint4 q = pack_bn_relu(acc[ot], sn2, tn2, cb);
    *(uint4*)&act[(size_t)(w * 16 + n) * ACT2_STRIDE + cb] = q;
  }
  __syncthreads();

  v8f acc2[8];
#pragma unroll
  for (int ot = 0; ot < 8; ++ot) acc2[ot] = zero;
#pragma unroll
  for (int kc = 0; kc < 4; ++kc) {
    v16bf bf = load_b_frag(act, ACT2_STRIDE, w, kc, lane);
#pragma unroll
    for (int ot = 0; ot < 8; ++ot) {
      v16bf af = load_a_frag(nw2bf, OUT_DIM, ot * 16, kc, lane);
      acc2[ot] = __builtin_amdgcn_wmma_f32_16x16x32_bf16(false, af, false, bf,
                                                         (short)0, acc2[ot], false, false);
    }
  }

  int p = w * 16 + n;
  float mk = mask[(size_t)b * Pn + p] ? 1.f : 0.f;
#pragma unroll
  for (int ot = 0; ot < 8; ++ot) {
#pragma unroll
    for (int v = 0; v < 8; ++v) {
      int o = ot * 16 + half * 8 + v;
      out[(size_t)b * OUT_DIM * Pn + (size_t)o * Pn + p] = acc2[ot][v] * mk;
    }
  }
}

// ============================ launch ============================
extern "C" void kernel_launch(void* const* d_in, const int* in_sizes, int n_in,
                              void* d_out, int out_size, void* d_ws, size_t ws_size,
                              hipStream_t stream) {
  (void)in_sizes; (void)n_in; (void)out_size; (void)ws_size;
  const float* pts = (const float*)d_in[0];
  const float* fts = (const float*)d_in[1];
  const float* lvs = (const float*)d_in[2];
  const unsigned char* mask = (const unsigned char*)d_in[3];   // jnp bool -> 1 byte
  const float* e_g1 = (const float*)d_in[4];
  const float* e_b1 = (const float*)d_in[5];
  const float* e_m1 = (const float*)d_in[6];
  const float* e_v1 = (const float*)d_in[7];
  const float* e_w1 = (const float*)d_in[8];
  const float* e_g2 = (const float*)d_in[9];
  const float* e_b2 = (const float*)d_in[10];
  const float* e_m2 = (const float*)d_in[11];
  const float* e_v2 = (const float*)d_in[12];
  const float* e_w2 = (const float*)d_in[13];
  const float* n_g1 = (const float*)d_in[14];
  const float* n_b1 = (const float*)d_in[15];
  const float* n_m1 = (const float*)d_in[16];
  const float* n_v1 = (const float*)d_in[17];
  const float* n_w1 = (const float*)d_in[18];
  const float* n_g2 = (const float*)d_in[19];
  const float* n_b2 = (const float*)d_in[20];
  const float* n_m2 = (const float*)d_in[21];
  const float* n_v2 = (const float*)d_in[22];
  const float* n_w2 = (const float*)d_in[23];

  char* ws = (char*)d_ws;
  int*   idx   = (int*)(ws + OFF_IDX);
  float* msg   = (float*)(ws + OFF_MSG);
  u16t*  w1bf  = (u16t*)(ws + OFF_W1);
  u16t*  w2bf  = (u16t*)(ws + OFF_W2);
  u16t*  nw1bf = (u16t*)(ws + OFF_NW1);
  u16t*  nw2bf = (u16t*)(ws + OFF_NW2);
  float* scb   = (float*)(ws + OFF_SCB);

  prep_kernel<<<1, 256, 0, stream>>>(e_g1, e_b1, e_m1, e_v1, e_w1,
                                     e_g2, e_b2, e_m2, e_v2, e_w2,
                                     n_g1, n_b1, n_m1, n_v1, n_w1,
                                     n_g2, n_b2, n_m2, n_v2, n_w2,
                                     w1bf, w2bf, nw1bf, nw2bf, scb);
  knn_kernel<<<(Bsz * Pn) / 8, 256, 0, stream>>>(pts, mask, idx);
  edge_mlp_kernel<<<Bsz * (Pn / 8), 256, 0, stream>>>(lvs, fts, mask, idx,
                                                      w1bf, w2bf, scb, msg);
  node_mlp_kernel<<<Bsz, 256, 0, stream>>>(msg, mask, nw1bf, nw2bf, scb,
                                           (float*)d_out);
}
